// CourierEncoder_58849641890595
// MI455X (gfx1250) — matrix operand order repaired and checked by reading, not compiled, not run on hardware
//
#include <hip/hip_runtime.h>
#include <hip/hip_fp16.h>
#include <stdint.h>

// ---------------- problem constants ----------------
#define B_ROWS   262144
#define PED      256
#define NED      128
#define CED      256
#define KDIM1    (PED + NED)        // 384
#define KT1      12                 // 384 / 32
#define KT2      8                  // 256 / 32
#define NT       16                 // 256 / 16
#define NEG_SLOPE 0.01f

// fragment-linear packed weight sizes (halves)
#define W1F_HALVES (KDIM1 * CED)    // 98304
#define W2F_HALVES (CED * CED)      // 65536

// LDS layout
#define WREG_BYTES (W1F_HALVES * 2)          // 196608 (w1 region, reused for w2)
#define H0_STRIDE  392                       // 384 + 8 halves (bank-conflict pad)
#define H1_STRIDE  264                       // 256 + 8 halves
#define H0_BYTES_PER_WAVE (16 * H0_STRIDE * 2) // 12544
#define NWAVES 8
#define LDS_BYTES (WREG_BYTES + NWAVES * H0_BYTES_PER_WAVE) // 296960 < 320KB

typedef __attribute__((ext_vector_type(16))) _Float16 v16h;
typedef __attribute__((ext_vector_type(8)))  float    v8f;

union Frag { v16h h; uint4 q[2]; };

// leaky_relu(x) == max(x, slope*x) exactly, for 0 < slope < 1:
// x>=0 -> x >= slope*x ; x<0 -> slope*x > x.  One v_mul + one v_max_num_f32.
__device__ __forceinline__ float leaky(float v) {
  return fmaxf(v, NEG_SLOPE * v);
}

// -------- prep: f32 weights -> f16, WMMA-B-fragment-linear layout in d_ws --------
// Within a 32(K)x16(N) tile: lane = 16*(kk>>4) + nn holds 16 contiguous halves,
// element e = kk&15  (contiguous k order, same order as the A fragments).
__global__ void prep_weights(const float* __restrict__ w1,
                             const float* __restrict__ w2,
                             _Float16* __restrict__ dst) {
  int i = blockIdx.x * blockDim.x + threadIdx.x;
  if (i < W1F_HALVES) {
    int k = i / CED, n = i % CED;
    int kt = k >> 5, kk = k & 31, nt = n >> 4, nn = n & 15;
    int off = ((nt * KT1 + kt) << 9) + ((((kk >> 4) << 4) + nn) << 4) + (kk & 15);
    dst[off] = (_Float16)w1[i];
  } else if (i < W1F_HALVES + W2F_HALVES) {
    int j = i - W1F_HALVES;
    int k = j / CED, n = j % CED;
    int kt = k >> 5, kk = k & 31, nt = n >> 4, nn = n & 15;
    int off = W1F_HALVES + ((nt * KT2 + kt) << 9) + ((((kk >> 4) << 4) + nn) << 4) + (kk & 15);
    dst[off] = (_Float16)w2[j];
  }
}

// async global -> LDS copy of one 16B chunk (CDNA5 async path, ASYNCcnt)
__device__ __forceinline__ void async_copy16(uint32_t lds_byte_addr,
                                             const char* gbase,
                                             uint32_t byte_off) {
  asm volatile("global_load_async_to_lds_b128 %0, %1, %2"
               :: "v"(lds_byte_addr), "v"(byte_off), "s"(gbase)
               : "memory");
}
__device__ __forceinline__ void wait_async0() {
  asm volatile("s_wait_asynccnt 0" ::: "memory");
}

__device__ __forceinline__ void load_bfrag(Frag& f, const _Float16* p) {
  f.q[0] = *(const uint4*)(p);
  f.q[1] = *(const uint4*)(p + 8);
}

// ---------------- fused encoder + 2-layer MLP ----------------
__global__ __launch_bounds__(256, 1)
void fused_mlp(const float* __restrict__ xy,   const float* __restrict__ tv,
               const float* __restrict__ w_sx, const float* __restrict__ b_sx,
               const float* __restrict__ w_cx, const float* __restrict__ b_cx,
               const float* __restrict__ w_sy, const float* __restrict__ b_sy,
               const float* __restrict__ w_cy, const float* __restrict__ b_cy,
               const float* __restrict__ w_t,  const float* __restrict__ b_t,
               const float* __restrict__ b1,   const float* __restrict__ b2,
               const _Float16* __restrict__ wfrag,
               float* __restrict__ out) {
  extern __shared__ char smem[];
  _Float16* wsh = (_Float16*)smem;                         // weight region (w1 then w2)
  const int tid  = threadIdx.x;
  const int wave = tid >> 5;
  const int lane = tid & 31;
  const int row  = lane & 15;       // M row within the wave's 16-row tile
  const int hsel = lane >> 4;       // lane half (A-frag K group / D row group)
  const int nlane = lane & 15;      // N column within an N-tile
  _Float16* hsh = (_Float16*)(smem + WREG_BYTES + wave * H0_BYTES_PER_WAVE);

  const uint32_t wsh_lds = (uint32_t)(uintptr_t)wsh;       // flat lo32 == LDS offset

  // ---- issue async staging of W1 fragments (overlaps with embedding math) ----
  {
    const char* g = (const char*)wfrag;
#pragma unroll 4
    for (int it = 0; it < (W1F_HALVES * 2 / 16) / 256; ++it) {
      uint32_t off = (uint32_t)(tid + it * 256) * 16u;
      async_copy16(wsh_lds + off, g, off);
    }
  }

  // ---- Fourier / time embedding into per-wave LDS tile (16 x 384, padded) ----
  const long grow = (long)blockIdx.x * 128 + wave * 16 + row;
  {
    const float x  = xy[grow * 2 + 0];
    const float y  = xy[grow * 2 + 1];
    const float tt = tv[grow];
    _Float16* hrow = hsh + row * H0_STRIDE;
    if (hsel == 0) {
      for (int j = 0; j < 64; ++j) hrow[j]       = (_Float16)__sinf(fmaf(x, w_sx[j], b_sx[j]));
      for (int j = 0; j < 64; ++j) hrow[64 + j]  = (_Float16)__cosf(fmaf(x, w_cx[j], b_cx[j]));
      for (int j = 0; j < 64; ++j) hrow[128 + j] = (_Float16)__sinf(fmaf(y, w_sy[j], b_sy[j]));
    } else {
      for (int j = 0; j < 64; ++j) hrow[192 + j] = (_Float16)__cosf(fmaf(y, w_cy[j], b_cy[j]));
      for (int j = 0; j < 128; ++j) hrow[256 + j] = (_Float16)leaky(fmaf(tt, w_t[j], b_t[j]));
    }
  }

  wait_async0();
  __syncthreads();

  // ---- preload GEMM1 A fragments (contiguous k = 16*hsel + e, same order as the B packing) ----
  Frag a1[KT1];
  {
    const _Float16* rb = hsh + row * H0_STRIDE + 16 * hsel;
#pragma unroll
    for (int kt = 0; kt < KT1; ++kt) {
      a1[kt].q[0] = *(const uint4*)(rb + kt * 32);
      a1[kt].q[1] = *(const uint4*)(rb + kt * 32 + 8);
    }
  }

  // ---- GEMM1: h1 = leaky(h0 @ W1 + b1), h1 -> per-wave LDS (reuses h0 region) ----
  // Two independent accumulator chains + double-buffered B fragments; bias folded
  // into the epilogue so its global load overlaps the whole WMMA stream.
  for (int nt = 0; nt < NT; ++nt) {
    const float bias = b1[nt * 16 + nlane];
    const _Float16* btile = wsh + (size_t)(nt * KT1) * 512 + lane * 16;
    Frag b0, b1f;
    load_bfrag(b0,  btile);
    load_bfrag(b1f, btile + 512);
    v8f acc0 = {};
    v8f acc1 = {};
#pragma unroll
    for (int kt = 0; kt < KT1; kt += 2) {
      Frag n0, n1;
      if (kt + 2 < KT1) load_bfrag(n0, btile + (size_t)(kt + 2) * 512);
      if (kt + 3 < KT1) load_bfrag(n1, btile + (size_t)(kt + 3) * 512);
      acc0 = __builtin_amdgcn_wmma_f32_16x16x32_f16(false, a1[kt].h,     false, b0.h,
                                                    (short)0, acc0, false, false);
      acc1 = __builtin_amdgcn_wmma_f32_16x16x32_f16(false, a1[kt + 1].h, false, b1f.h,
                                                    (short)0, acc1, false, false);
      b0 = n0; b1f = n1;
    }
#pragma unroll
    for (int v = 0; v < 8; ++v) {
      float val = leaky(acc0[v] + acc1[v] + bias);
      hsh[(v + 8 * hsel) * H1_STRIDE + nt * 16 + nlane] = (_Float16)val;
    }
  }

  __syncthreads();   // all waves done reading W1 from LDS

  // ---- async-stage W2 fragments into same LDS region ----
  {
    const char* g = (const char*)wfrag + (size_t)W1F_HALVES * 2;
#pragma unroll 4
    for (int it = 0; it < (W2F_HALVES * 2 / 16) / 256; ++it) {
      uint32_t off = (uint32_t)(tid + it * 256) * 16u;
      async_copy16(wsh_lds + off, g, off);
    }
  }

  // ---- preload GEMM2 A fragments from h1 (independent LDS region) ----
  Frag a2[KT2];
  {
    const _Float16* rb = hsh + row * H1_STRIDE + 16 * hsel;
#pragma unroll
    for (int kt = 0; kt < KT2; ++kt) {
      a2[kt].q[0] = *(const uint4*)(rb + kt * 32);
      a2[kt].q[1] = *(const uint4*)(rb + kt * 32 + 8);
    }
  }

  wait_async0();
  __syncthreads();

  // ---- GEMM2: out = leaky(h1 @ W2 + b2) -> global f32 ----
  float* orow = out + ((long)blockIdx.x * 128 + wave * 16) * CED;
  for (int nt = 0; nt < NT; ++nt) {
    const float bias = b2[nt * 16 + nlane];
    const _Float16* btile = wsh + (size_t)(nt * KT2) * 512 + lane * 16;
    Frag b0, b1f;
    load_bfrag(b0,  btile);
    load_bfrag(b1f, btile + 512);
    v8f acc0 = {};
    v8f acc1 = {};
#pragma unroll
    for (int kt = 0; kt < KT2; kt += 2) {
      Frag n0, n1;
      if (kt + 2 < KT2) load_bfrag(n0, btile + (size_t)(kt + 2) * 512);
      if (kt + 3 < KT2) load_bfrag(n1, btile + (size_t)(kt + 3) * 512);
      acc0 = __builtin_amdgcn_wmma_f32_16x16x32_f16(false, a2[kt].h,     false, b0.h,
                                                    (short)0, acc0, false, false);
      acc1 = __builtin_amdgcn_wmma_f32_16x16x32_f16(false, a2[kt + 1].h, false, b1f.h,
                                                    (short)0, acc1, false, false);
      b0 = n0; b1f = n1;
    }
#pragma unroll
    for (int v = 0; v < 8; ++v) {
      orow[(v + 8 * hsel) * CED + nt * 16 + nlane] = leaky(acc0[v] + acc1[v] + bias);
    }
  }
}

extern "C" void kernel_launch(void* const* d_in, const int* in_sizes, int n_in,
                              void* d_out, int out_size, void* d_ws, size_t ws_size,
                              hipStream_t stream) {
  const float* xy   = (const float*)d_in[0];
  const float* t    = (const float*)d_in[1];
  const float* w_sx = (const float*)d_in[2];
  const float* b_sx = (const float*)d_in[3];
  const float* w_cx = (const float*)d_in[4];
  const float* b_cx = (const float*)d_in[5];
  const float* w_sy = (const float*)d_in[6];
  const float* b_sy = (const float*)d_in[7];
  const float* w_cy = (const float*)d_in[8];
  const float* b_cy = (const float*)d_in[9];
  const float* w_t  = (const float*)d_in[10];
  const float* b_t  = (const float*)d_in[11];
  const float* w1   = (const float*)d_in[12];
  const float* b1   = (const float*)d_in[13];
  const float* w2   = (const float*)d_in[14];
  const float* b2   = (const float*)d_in[15];
  float* out = (float*)d_out;
  _Float16* wfrag = (_Float16*)d_ws;

  const int totalW = W1F_HALVES + W2F_HALVES;
  prep_weights<<<(totalW + 255) / 256, 256, 0, stream>>>(w1, w2, wfrag);

  hipFuncSetAttribute((const void*)fused_mlp,
                      hipFuncAttributeMaxDynamicSharedMemorySize, LDS_BYTES);

  fused_mlp<<<B_ROWS / 128, 256, LDS_BYTES, stream>>>(
      xy, t, w_sx, b_sx, w_cx, b_cx, w_sy, b_sy, w_cy, b_cy,
      w_t, b_t, b1, b2, wfrag, out);
}
